// QuantizedLinear_76192719831745
// MI455X (gfx1250) — 8-bit matrix operand order repaired and checked by reading, compiled, not run on hardware
//
#include <hip/hip_runtime.h>

#define TOKENS 8192
#define DIN    4096
#define DOUT   4096

typedef int v8i __attribute__((ext_vector_type(8)));

__global__ void __launch_bounds__(256)
quant_rows_kernel(const float* __restrict__ src,
                  const float* __restrict__ scales,
                  const int*   __restrict__ zps,
                  int perRow,
                  unsigned char* __restrict__ q,
                  int* __restrict__ rowsum)
{
    const int row = blockIdx.x;
    const int tid = threadIdx.x;

    const int   pidx = perRow ? row : 0;
    const float s    = scales[pidx];
    const float inv  = 1.0f / s;
    const float zpf  = (float)zps[pidx];

    const size_t base = (size_t)row * DIN + (size_t)tid * 16;

    unsigned int packed[4];
    int sum = 0;
#pragma unroll
    for (int g = 0; g < 4; ++g) {
        float4 v = *(const float4*)(src + base + g * 4);
        float q0 = fminf(fmaxf(rintf(v.x * inv) + zpf, 0.0f), 255.0f);
        float q1 = fminf(fmaxf(rintf(v.y * inv) + zpf, 0.0f), 255.0f);
        float q2 = fminf(fmaxf(rintf(v.z * inv) + zpf, 0.0f), 255.0f);
        float q3 = fminf(fmaxf(rintf(v.w * inv) + zpf, 0.0f), 255.0f);
        int i0 = (int)q0, i1 = (int)q1, i2 = (int)q2, i3 = (int)q3;
        packed[g] = (unsigned)i0 | ((unsigned)i1 << 8) |
                    ((unsigned)i2 << 16) | ((unsigned)i3 << 24);
        sum += i0 + i1 + i2 + i3;
    }
    uint4 out;
    out.x = packed[0]; out.y = packed[1]; out.z = packed[2]; out.w = packed[3];
    *(uint4*)(q + base) = out;

    __shared__ int sdata[256];
    sdata[tid] = sum;
    __syncthreads();
#pragma unroll
    for (int sft = 128; sft > 0; sft >>= 1) {
        if (tid < sft) sdata[tid] += sdata[tid + sft];
        __syncthreads();
    }
    if (tid == 0) rowsum[row] = sdata[0];
}

__global__ void __launch_bounds__(256)
qgemm_wmma_kernel(const unsigned char* __restrict__ qx,
                  const unsigned char* __restrict__ qw,
                  const int*   __restrict__ rsx,
                  const int*   __restrict__ rsw,
                  const float* __restrict__ bias,
                  const float* __restrict__ act_scale,
                  const int*   __restrict__ act_zp,
                  const float* __restrict__ w_scale,
                  const int*   __restrict__ w_zp,
                  float* __restrict__ out)
{
    __shared__ __align__(16) unsigned char lds_a[2][128 * 64];
    __shared__ __align__(16) unsigned char lds_b[2][128 * 64];

    const int tid    = threadIdx.x;
    const int lane   = tid & 31;
    const int wave   = tid >> 5;
    const int waveM  = wave & 3;
    const int waveN  = wave >> 2;
    const int m      = lane & 15;
    const int h      = lane >> 4;

    const int blockM = blockIdx.y * 128;
    const int blockN = blockIdx.x * 128;

    const int srow  = tid >> 1;
    const int shalf = tid & 1;
    const unsigned char* aSrc = qx + (size_t)(blockM + srow) * DIN + shalf * 32;
    const unsigned char* bSrc = qw + (size_t)(blockN + srow) * DIN + shalf * 32;
    const int ldsOff = srow * 64 + shalf * 32;

    const v8i vzero = {0, 0, 0, 0, 0, 0, 0, 0};
    v8i acc[2][4];
#pragma unroll
    for (int mt = 0; mt < 2; ++mt)
#pragma unroll
        for (int nt = 0; nt < 4; ++nt) acc[mt][nt] = vzero;

    {
        const uint4* ga = (const uint4*)aSrc;
        const uint4* gb = (const uint4*)bSrc;
        uint4 a0 = ga[0], a1 = ga[1];
        uint4 b0 = gb[0], b1 = gb[1];
        uint4* la = (uint4*)&lds_a[0][ldsOff];
        uint4* lb = (uint4*)&lds_b[0][ldsOff];
        la[0] = a0; la[1] = a1;
        lb[0] = b0; lb[1] = b1;
    }
    __syncthreads();

    int p = 0;
    for (int k0 = 0; k0 < DIN; k0 += 64) {
        const bool hasNext = (k0 + 64) < DIN;

        uint4 na0 = {}, na1 = {}, nb0 = {}, nb1 = {};
        if (hasNext) {
            const uint4* ga = (const uint4*)(aSrc + k0 + 64);
            const uint4* gb = (const uint4*)(bSrc + k0 + 64);
            na0 = ga[0]; na1 = ga[1];
            nb0 = gb[0]; nb1 = gb[1];
            if (k0 + 128 < DIN) {
                __builtin_prefetch(aSrc + k0 + 128, 0, 3);
                __builtin_prefetch(bSrc + k0 + 128, 0, 3);
            }
        }

        v8i af[2];
#pragma unroll
        for (int mt = 0; mt < 2; ++mt) {
            const unsigned char* arow =
                &lds_a[p][(waveM * 32 + mt * 16 + m) * 64];
#pragma unroll
            for (int j = 0; j < 4; ++j) {
                int2 pr = *(const int2*)(arow + h * 16 + (j & 1) * 8 + (j >> 1) * 32);
                af[mt][2 * j]     = pr.x;
                af[mt][2 * j + 1] = pr.y;
            }
        }

        v8i bf[4];
#pragma unroll
        for (int nt = 0; nt < 4; ++nt) {
            const unsigned char* brow =
                &lds_b[p][(waveN * 64 + nt * 16 + m) * 64];
            int4 p0 = *(const int4*)(brow + h * 16);
            int4 p1 = *(const int4*)(brow + h * 16 + 32);
            bf[nt][0] = p0.x; bf[nt][1] = p0.y; bf[nt][2] = p0.z; bf[nt][3] = p0.w;
            bf[nt][4] = p1.x; bf[nt][5] = p1.y; bf[nt][6] = p1.z; bf[nt][7] = p1.w;
        }

#pragma unroll
        for (int mt = 0; mt < 2; ++mt)
#pragma unroll
            for (int nt = 0; nt < 4; ++nt)
                acc[mt][nt] = __builtin_amdgcn_wmma_i32_16x16x64_iu8(
                     false, af[mt],
                     false, bf[nt],
                    acc[mt][nt],
                     false,  false);

        if (hasNext) {
            uint4* la = (uint4*)&lds_a[p ^ 1][ldsOff];
            uint4* lb = (uint4*)&lds_b[p ^ 1][ldsOff];
            la[0] = na0; la[1] = na1;
            lb[0] = nb0; lb[1] = nb1;
        }
        __syncthreads();
        p ^= 1;
    }

    const float as   = act_scale[0];
    const int   zpaI = act_zp[0];

#pragma unroll
    for (int nt = 0; nt < 4; ++nt) {
        const int col = blockN + waveN * 64 + nt * 16 + m;
        const float wsc  = w_scale[col];
        const float bcol = bias[col];
        const int   zpwI = w_zp[col];
        const int cN = DIN * zpaI * zpwI - zpaI * rsw[col];
        const float sc = as * wsc;
#pragma unroll
        for (int mt = 0; mt < 2; ++mt) {
            const int rowBase = blockM + waveM * 32 + mt * 16 + h * 8;
#pragma unroll
            for (int r = 0; r < 8; ++r) {
                const int rowr = rowBase + r;
                const int I = acc[mt][nt][r] - zpwI * rsx[rowr] + cN;
                out[(size_t)rowr * DOUT + col] = sc * (float)I + bcol;
            }
        }
    }
}

extern "C" void kernel_launch(void* const* d_in, const int* in_sizes, int n_in,
                              void* d_out, int out_size, void* d_ws, size_t ws_size,
                              hipStream_t stream) {
    const float* x    = (const float*)d_in[0];
    const float* w    = (const float*)d_in[1];
    const float* b    = (const float*)d_in[2];
    const float* a_s  = (const float*)d_in[3];
    const int*   a_zp = (const int*)  d_in[4];
    const float* w_s  = (const float*)d_in[5];
    const int*   w_zp = (const int*)  d_in[6];
    float* out = (float*)d_out;

    unsigned char* ws = (unsigned char*)d_ws;
    unsigned char* qx  = ws;
    unsigned char* qw  = qx + (size_t)TOKENS * DIN;
    int*           rsx = (int*)(qw + (size_t)DOUT * DIN);
    int*           rsw = rsx + TOKENS;

    quant_rows_kernel<<<TOKENS, 256, 0, stream>>>(x, a_s, a_zp, 0, qx, rsx);
    quant_rows_kernel<<<DOUT,   256, 0, stream>>>(w, w_s, w_zp, 1, qw, rsw);

    dim3 grid(DOUT / 128, TOKENS / 128);
    qgemm_wmma_kernel<<<grid, 256, 0, stream>>>(
        qx, qw, rsx, rsw, b, a_s, a_zp, w_s, w_zp, out);
}
